// ImaginaryContext_34385508172407
// MI455X (gfx1250) — matrix operand order repaired and checked by reading, not compiled, not run on hardware
//
#include <hip/hip_runtime.h>

// ---------------------------------------------------------------------------
// ImaginaryContext for MI455X (gfx1250): bf16 WMMA everywhere, f32 accum.
// B=256, N=512, DH=DM=256, K=25, H=4, dph=64, S=N+K=537.
// ---------------------------------------------------------------------------

#define NB   256      // batch
#define NCTX 512      // context rows
#define DH_  256
#define KSL  25       // imaginary slots
#define SEQ  537      // NCTX + KSL
#define RSZ  (NB * KSL * DH_)   // 1,638,400 elems per output tensor

typedef __attribute__((ext_vector_type(16))) __bf16 v16bf;
typedef __attribute__((ext_vector_type(8)))  __bf16 v8bf;
typedef __attribute__((ext_vector_type(4)))  __bf16 v4bf;
typedef __attribute__((ext_vector_type(8)))  float  v8f;
typedef __attribute__((ext_vector_type(4)))  float  v4f;

__device__ __forceinline__ v8f wmma_bf(v16bf a, v16bf b, v8f c) {
  // D = A(16x32 bf16) * B(32x16 bf16) + C(16x16 f32)
  return __builtin_amdgcn_wmma_f32_16x16x32_bf16(false, a, false, b, (short)0, c,
                                                 false, false);
}

__device__ __forceinline__ v8f zero8() {
  v8f z = {0.f, 0.f, 0.f, 0.f, 0.f, 0.f, 0.f, 0.f};
  return z;
}

// A-operand (16x32, MxK) loaded from f32 row-major, converted to bf16.
// lane L: m = m0 + (L&15), half = L>>4; elems e<8: k=k0+8*half+e;
// e>=8: k=k0+16+8*half+(e-8).
__device__ __forceinline__ v16bf load_A_f32(const float* A, int lda, int m0,
                                            int k0, int lane) {
  int m = m0 + (lane & 15), half = lane >> 4;
  const float* p = A + (size_t)m * lda + k0 + 8 * half;
  v4f x0 = *(const v4f*)(p);
  v4f x1 = *(const v4f*)(p + 4);
  v4f x2 = *(const v4f*)(p + 16);
  v4f x3 = *(const v4f*)(p + 20);
  v16bf r;
#pragma unroll
  for (int e = 0; e < 4; ++e) {
    r[e]      = (__bf16)x0[e];
    r[4 + e]  = (__bf16)x1[e];
    r[8 + e]  = (__bf16)x2[e];
    r[12 + e] = (__bf16)x3[e];
  }
  return r;
}

// A-operand from bf16 row-major (LDS or global).
__device__ __forceinline__ v16bf load_A_bf(const __bf16* A, int lda, int m0,
                                           int k0, int lane) {
  int m = m0 + (lane & 15), half = lane >> 4;
  const __bf16* p = A + (size_t)m * lda + k0 + 8 * half;
  v8bf lo = *(const v8bf*)(p);
  v8bf hi = *(const v8bf*)(p + 16);
  return __builtin_shufflevector(lo, hi, 0, 1, 2, 3, 4, 5, 6, 7, 8, 9, 10, 11,
                                 12, 13, 14, 15);
}

// B-operand (32x16, KxN) from row-major weight W[n][k] (i.e. computes X @ W^T).
// lane L: n = n0 + (L&15), half = L>>4; elems e<8: k=k0+8*half+e;
// e>=8: k=k0+16+8*half+(e-8).
__device__ __forceinline__ v16bf load_B_bf(const __bf16* W, int ldb, int n0,
                                           int k0, int lane) {
  int n = n0 + (lane & 15), half = lane >> 4;
  const __bf16* p = W + (size_t)n * ldb + k0 + 8 * half;
  v8bf lo = *(const v8bf*)(p);
  v8bf hi = *(const v8bf*)(p + 16);
  return __builtin_shufflevector(lo, hi, 0, 1, 2, 3, 4, 5, 6, 7, 8, 9, 10, 11,
                                 12, 13, 14, 15);
}

__device__ __forceinline__ float sigmoidf_(float x) {
  return 1.f / (1.f + __expf(-x));
}

// ---------------------------------------------------------------------------
// 0) f32 -> bf16 (sub)matrix convert:  dst[r*cols+c] = src[r*src_ld+col0+c]
// ---------------------------------------------------------------------------
__global__ void cvt_kernel(const float* __restrict__ src,
                           __bf16* __restrict__ dst, int rows, int src_ld,
                           int col0, int cols) {
  int i = blockIdx.x * blockDim.x + threadIdx.x;
  if (i >= rows * cols) return;
  int r = i / cols, c = i - r * cols;
  dst[i] = (__bf16)src[(size_t)r * src_ld + col0 + c];
}

// ---------------------------------------------------------------------------
// 1) mean over context:  mean_v[b, d] = mean_n v_re[b, n, d]
// ---------------------------------------------------------------------------
__global__ void mean_v_kernel(const float* __restrict__ v,
                              float* __restrict__ out) {
  int b = blockIdx.x, d = threadIdx.x;
  const float* p = v + (size_t)b * NCTX * DH_ + d;
  float s = 0.f;
  for (int n = 0; n < NCTX; ++n) s += p[(size_t)n * DH_];
  out[b * DH_ + d] = s * (1.f / (float)NCTX);
}

// ---------------------------------------------------------------------------
// 2) LSTM gates with zero initial state (f-gate dead, Whh dead):
//    h = sigmoid(o) * tanh( sigmoid(i) * tanh(g) ),  gates = A @ Wg^T + bih+bhh
//    A: [M,256] f32, Wg: [1024,256] bf16 (gate rows i|f|g|o), H out: [M,256] f32
// ---------------------------------------------------------------------------
__global__ __launch_bounds__(256) void lstm_gates_kernel(
    const float* __restrict__ A, const __bf16* __restrict__ Wg,
    const float* __restrict__ bih, const float* __restrict__ bhh,
    float* __restrict__ H) {
  int m0 = blockIdx.x * 16;
  int wv = threadIdx.x >> 5, lane = threadIdx.x & 31;
  v16bf a[8];
#pragma unroll
  for (int kk = 0; kk < 8; ++kk) a[kk] = load_A_f32(A, DH_, m0, kk * 32, lane);
  for (int jj = 0; jj < 2; ++jj) {
    int j = wv + jj * 8;  // column tile 0..15 within 256 gate cols
    v8f ai = zero8(), ag = zero8(), ao = zero8();
#pragma unroll
    for (int kk = 0; kk < 8; ++kk) {
      ai = wmma_bf(a[kk], load_B_bf(Wg, DH_, 0 * DH_ + j * 16, kk * 32, lane), ai);
      ag = wmma_bf(a[kk], load_B_bf(Wg, DH_, 2 * DH_ + j * 16, kk * 32, lane), ag);
      ao = wmma_bf(a[kk], load_B_bf(Wg, DH_, 3 * DH_ + j * 16, kk * 32, lane), ao);
    }
    int n = j * 16 + (lane & 15), half = lane >> 4;
    float bi = bih[n] + bhh[n];
    float bg = bih[2 * DH_ + n] + bhh[2 * DH_ + n];
    float bo = bih[3 * DH_ + n] + bhh[3 * DH_ + n];
#pragma unroll
    for (int r = 0; r < 8; ++r) {
      int row = m0 + r + 8 * half;
      float cc = sigmoidf_(ai[r] + bi) * tanhf(ag[r] + bg);
      H[(size_t)row * DH_ + n] = sigmoidf_(ao[r] + bo) * tanhf(cc);
    }
  }
}

// ---------------------------------------------------------------------------
// 3) out = relu(A @ W^T + b)   (M=256, 256x256)
// ---------------------------------------------------------------------------
__global__ __launch_bounds__(256) void gemm_relu_kernel(
    const float* __restrict__ A, const __bf16* __restrict__ W,
    const float* __restrict__ bias, float* __restrict__ out) {
  int m0 = blockIdx.x * 16;
  int wv = threadIdx.x >> 5, lane = threadIdx.x & 31;
  v16bf a[8];
#pragma unroll
  for (int kk = 0; kk < 8; ++kk) a[kk] = load_A_f32(A, DH_, m0, kk * 32, lane);
  for (int jj = 0; jj < 2; ++jj) {
    int j = wv + jj * 8;
    v8f acc = zero8();
#pragma unroll
    for (int kk = 0; kk < 8; ++kk)
      acc = wmma_bf(a[kk], load_B_bf(W, DH_, j * 16, kk * 32, lane), acc);
    int n = j * 16 + (lane & 15), half = lane >> 4;
    float bv = bias[n];
#pragma unroll
    for (int r = 0; r < 8; ++r)
      out[(size_t)(m0 + r + 8 * half) * DH_ + n] = fmaxf(acc[r] + bv, 0.f);
  }
}

// ---------------------------------------------------------------------------
// 4) ikey heads: mu = A@Wmu^T + bmu ; sig = exp(0.5*(A@Wls^T + bls))
// ---------------------------------------------------------------------------
__global__ __launch_bounds__(256) void ikey_heads_kernel(
    const float* __restrict__ A, const __bf16* __restrict__ Wmu,
    const __bf16* __restrict__ Wls, const float* __restrict__ bmu,
    const float* __restrict__ bls, float* __restrict__ muo,
    float* __restrict__ sigo) {
  int m0 = blockIdx.x * 16;
  int wv = threadIdx.x >> 5, lane = threadIdx.x & 31;
  v16bf a[8];
#pragma unroll
  for (int kk = 0; kk < 8; ++kk) a[kk] = load_A_f32(A, DH_, m0, kk * 32, lane);
  for (int jj = 0; jj < 2; ++jj) {
    int j = wv + jj * 8;
    v8f am = zero8(), al = zero8();
#pragma unroll
    for (int kk = 0; kk < 8; ++kk) {
      am = wmma_bf(a[kk], load_B_bf(Wmu, DH_, j * 16, kk * 32, lane), am);
      al = wmma_bf(a[kk], load_B_bf(Wls, DH_, j * 16, kk * 32, lane), al);
    }
    int n = j * 16 + (lane & 15), half = lane >> 4;
    float bm = bmu[n], bl = bls[n];
#pragma unroll
    for (int r = 0; r < 8; ++r) {
      int row = m0 + r + 8 * half;
      muo[(size_t)row * DH_ + n] = am[r] + bm;
      sigo[(size_t)row * DH_ + n] = __expf(0.5f * (al[r] + bl));
    }
  }
}

// ---------------------------------------------------------------------------
// 5) x_im[b,k,d] = mu[b,d] + sig[b,d] * eps_x[b,k,d]
// ---------------------------------------------------------------------------
__global__ void xim_kernel(const float* __restrict__ mu,
                           const float* __restrict__ sig,
                           const float* __restrict__ eps,
                           float* __restrict__ xout) {
  int i = blockIdx.x * 256 + threadIdx.x;  // < RSZ
  int d = i & 255;
  int b = i / (KSL * DH_);
  xout[i] = mu[b * DH_ + d] + sig[b * DH_ + d] * eps[i];
}

// ---------------------------------------------------------------------------
// 6) Fused per-batch multi-head flash attention with inline K/V projection,
//    output projection (Wf, 512->256), residual and LayerNorm.
//    One workgroup (8 waves) per batch; wave = (head h, q-row-tile qt).
//    LDS (64 KB): [0,16K)=qb  [16K,32K)=stage/scores/ao  [32K,48K)=kp/p/alpha/l
//                 [48K,64K)=vpT ; [32K,64K) reused as f32 pre-LN buffer.
// ---------------------------------------------------------------------------

// Stage one 32x256 tile into LDS as bf16 using float4 loads / v4bf stores.
// Rows gr<NCTX come from `re` (f32), NCTX<=gr<SEQ from `im` (f32), else zero.
__device__ __forceinline__ void stage_tile(__bf16* dst, const float* re,
                                           const float* im, int b, int r0,
                                           int tid) {
#pragma unroll
  for (int i = 0; i < 8; ++i) {
    int cidx = tid + i * 256;          // 0..2047 float4-chunks
    int row = cidx >> 6, c4 = (cidx & 63) * 4;
    int gr = r0 + row;
    v4f x;
    if (gr < NCTX)
      x = *(const v4f*)(re + (size_t)b * NCTX * DH_ + (size_t)gr * DH_ + c4);
    else if (gr < SEQ)
      x = *(const v4f*)(im + (size_t)b * KSL * DH_ +
                        (size_t)(gr - NCTX) * DH_ + c4);
    else
      x = (v4f){0.f, 0.f, 0.f, 0.f};
    v4bf y;
#pragma unroll
    for (int e = 0; e < 4; ++e) y[e] = (__bf16)x[e];
    *(v4bf*)(dst + row * DH_ + c4) = y;
  }
}

__global__ __launch_bounds__(256) void attn_kernel(
    const float* __restrict__ q_src,  // [B*25,256] f32
    const float* __restrict__ x_re,   // [B,512,256]
    const float* __restrict__ v_re,   // [B,512,256]
    const float* __restrict__ x_im,   // [B*25,256] f32
    const float* __restrict__ itr_h,  // [B*25,256] f32
    const __bf16* __restrict__ Wk, const __bf16* __restrict__ Wv,
    const __bf16* __restrict__ Wq, const __bf16* __restrict__ Wf,
    const float* __restrict__ bf_, const float* __restrict__ ln_g,
    const float* __restrict__ ln_b, float* __restrict__ q_out) {
  __shared__ __attribute__((aligned(128))) char smem[65536];
  __bf16* qb   = (__bf16*)(smem);            // [32][256] residual/q input (bf16)
  __bf16* kxb  = (__bf16*)(smem + 16384);    // staging [32][256] bf16
  __bf16* kpb  = (__bf16*)(smem + 32768);    // K-proj  [32][256] bf16
  __bf16* vpT  = (__bf16*)(smem + 49152);    // V-proj transposed [256][32] bf16
  float*  sbuf = (float*)(smem + 16384);     // scores [4][32][32] f32 (alias)
  __bf16* pbuf = (__bf16*)(smem + 32768);    // probs  [4][32][32] bf16 (alias)
  float*  alpha_s = (float*)(smem + 32768 + 8192);        // [128]
  float*  lsum_s  = (float*)(smem + 32768 + 8192 + 512);  // [128]
  __bf16* aob = (__bf16*)(smem + 16384);     // attn out [32][256] bf16 (alias)
  float*  o32 = (float*)(smem + 32768);      // pre-LN   [32][256] f32 (alias)

  int b = blockIdx.x;
  int tid = threadIdx.x, lane = tid & 31, wv = tid >> 5;
  int h = wv >> 1, qt = wv & 1;
  const float* qsrc_b = q_src + (size_t)b * KSL * DH_;

  // stage q (rows 25..31 zero)
#pragma unroll
  for (int i = 0; i < 8; ++i) {
    int cidx = tid + i * 256;
    int row = cidx >> 6, c4 = (cidx & 63) * 4;
    v4f x = (row < KSL) ? *(const v4f*)(qsrc_b + row * DH_ + c4)
                        : (v4f){0.f, 0.f, 0.f, 0.f};
    v4bf y;
#pragma unroll
    for (int e = 0; e < 4; ++e) y[e] = (__bf16)x[e];
    *(v4bf*)(qb + row * DH_ + c4) = y;
  }
  __syncthreads();

  // q projection -> kxb (temporary), then each wave grabs its head slice.
  for (int i = 0; i < 4; ++i) {
    int t = wv * 4 + i, rt = t >> 4, ct = t & 15;
    v8f acc = zero8();
#pragma unroll
    for (int kk = 0; kk < 8; ++kk)
      acc = wmma_bf(load_A_bf(qb, DH_, rt * 16, kk * 32, lane),
                    load_B_bf(Wq, DH_, ct * 16, kk * 32, lane), acc);
    int n = ct * 16 + (lane & 15), half = lane >> 4;
#pragma unroll
    for (int r = 0; r < 8; ++r)
      kxb[(rt * 16 + r + 8 * half) * DH_ + n] = (__bf16)acc[r];
  }
  __syncthreads();
  v16bf qpa0 = load_A_bf(kxb, DH_, qt * 16, h * 64 + 0, lane);
  v16bf qpa1 = load_A_bf(kxb, DH_, qt * 16, h * 64 + 32, lane);
  __syncthreads();

  float m_i = -1e30f, l_i = 0.f;  // online-softmax state (threads < 128)
  v8f oacc[4];
#pragma unroll
  for (int c = 0; c < 4; ++c) oacc[c] = zero8();

  for (int kt = 0; kt < 17; ++kt) {
    int r0 = kt * 32;
    // ---- prefetch next tile toward L2/L0 (global_prefetch_b8) ----
    {
      int r0n = r0 + 32;
#pragma unroll
      for (int i = 0; i < 2; ++i) {
        int line = tid + i * 256;          // 64B cache line within 32x1KB tile
        int row = r0n + (line >> 4);
        int cf = (line & 15) * 16;         // float index of line start
        if (row < NCTX) {
          const float* pk =
              x_re + (size_t)b * NCTX * DH_ + (size_t)row * DH_ + cf;
          const float* pv =
              v_re + (size_t)b * NCTX * DH_ + (size_t)row * DH_ + cf;
          __builtin_prefetch(pk, 0, 1);
          __builtin_prefetch(pv, 0, 1);
        }
      }
    }
    // ---- stage key tile (bf16) ----
    stage_tile(kxb, x_re, x_im, b, r0, tid);
    __syncthreads();
    // ---- K projection: kp[32][256] ----
    for (int i = 0; i < 4; ++i) {
      int t = wv * 4 + i, rt = t >> 4, ct = t & 15;
      v8f acc = zero8();
#pragma unroll
      for (int kk = 0; kk < 8; ++kk)
        acc = wmma_bf(load_A_bf(kxb, DH_, rt * 16, kk * 32, lane),
                      load_B_bf(Wk, DH_, ct * 16, kk * 32, lane), acc);
      int n = ct * 16 + (lane & 15), half = lane >> 4;
#pragma unroll
      for (int r = 0; r < 8; ++r)
        kpb[(rt * 16 + r + 8 * half) * DH_ + n] = (__bf16)acc[r];
    }
    __syncthreads();
    // ---- stage value tile ----
    stage_tile(kxb, v_re, itr_h, b, r0, tid);
    __syncthreads();
    // ---- V projection, stored transposed: vpT[col][krow] ----
    for (int i = 0; i < 4; ++i) {
      int t = wv * 4 + i, rt = t >> 4, ct = t & 15;
      v8f acc = zero8();
#pragma unroll
      for (int kk = 0; kk < 8; ++kk)
        acc = wmma_bf(load_A_bf(kxb, DH_, rt * 16, kk * 32, lane),
                      load_B_bf(Wv, DH_, ct * 16, kk * 32, lane), acc);
      int n = ct * 16 + (lane & 15), half = lane >> 4;
      __bf16* dst = vpT + n * 32 + rt * 16 + 8 * half;
#pragma unroll
      for (int r = 0; r < 8; ++r) dst[r] = (__bf16)acc[r];
    }
    __syncthreads();
    // ---- scores: wave (h,qt) computes s[16 q][32 k] ----
    for (int ct2 = 0; ct2 < 2; ++ct2) {
      v8f acc = zero8();
      acc = wmma_bf(qpa0, load_B_bf(kpb, DH_, ct2 * 16, h * 64 + 0, lane), acc);
      acc = wmma_bf(qpa1, load_B_bf(kpb, DH_, ct2 * 16, h * 64 + 32, lane), acc);
      int n = ct2 * 16 + (lane & 15), half = lane >> 4;
#pragma unroll
      for (int r = 0; r < 8; ++r) {
        int qrow = qt * 16 + r + 8 * half;
        sbuf[(h * 32 + qrow) * 32 + n] = acc[r] * 0.125f;  // 1/sqrt(64)
      }
    }
    __syncthreads();
    // ---- online softmax (threads 0..127 own one (head, q-row)) ----
    if (tid < 128) {
      int kmax = SEQ - r0; if (kmax > 32) kmax = 32;
      float* srow = sbuf + tid * 32;
      __bf16* prow = pbuf + tid * 32;
      float mx = m_i;
      for (int j = 0; j < kmax; ++j) mx = fmaxf(mx, srow[j]);
      float al = __expf(m_i - mx), sum = 0.f;
      for (int j = 0; j < 32; ++j) {
        float p = (j < kmax) ? __expf(srow[j] - mx) : 0.f;
        prow[j] = (__bf16)p;
        sum += p;
      }
      l_i = l_i * al + sum;
      m_i = mx;
      alpha_s[tid] = al;
    }
    __syncthreads();
    // ---- rescale + accumulate out += P @ Vp (per wave) ----
    {
      int half = lane >> 4;
      float av[8];
#pragma unroll
      for (int r = 0; r < 8; ++r)
        av[r] = alpha_s[h * 32 + qt * 16 + r + 8 * half];
      v16bf pa = load_A_bf(pbuf + (h * 32 + qt * 16) * 32, 32, 0, 0, lane);
#pragma unroll
      for (int c = 0; c < 4; ++c) {
        v8f acc = oacc[c];
#pragma unroll
        for (int r = 0; r < 8; ++r) acc[r] *= av[r];
        oacc[c] = wmma_bf(pa, load_B_bf(vpT, 32, h * 64 + c * 16, 0, lane), acc);
      }
    }
    __syncthreads();
  }

  // ---- finalize: out /= l, write ao (bf16) ----
  if (tid < 128) lsum_s[tid] = l_i;
  __syncthreads();
  {
    int half = lane >> 4;
    float lv[8];
#pragma unroll
    for (int r = 0; r < 8; ++r) {
      float d = lsum_s[h * 32 + qt * 16 + r + 8 * half];
      lv[r] = (d > 0.f) ? 1.f / d : 0.f;
    }
#pragma unroll
    for (int c = 0; c < 4; ++c) {
      int n = h * 64 + c * 16 + (lane & 15);
#pragma unroll
      for (int r = 0; r < 8; ++r)
        aob[(qt * 16 + r + 8 * half) * DH_ + n] = (__bf16)(oacc[c][r] * lv[r]);
    }
  }
  __syncthreads();

  // ---- output projection: o = [qb | ao] @ Wf^T + bf + resid ----
  for (int i = 0; i < 4; ++i) {
    int t = wv * 4 + i, rt = t >> 4, ct = t & 15;
    v8f acc = zero8();
#pragma unroll
    for (int kk = 0; kk < 8; ++kk)
      acc = wmma_bf(load_A_bf(qb, DH_, rt * 16, kk * 32, lane),
                    load_B_bf(Wf, 512, ct * 16, kk * 32, lane), acc);
#pragma unroll
    for (int kk = 0; kk < 8; ++kk)
      acc = wmma_bf(load_A_bf(aob, DH_, rt * 16, kk * 32, lane),
                    load_B_bf(Wf, 512, ct * 16, 256 + kk * 32, lane), acc);
    int n = ct * 16 + (lane & 15), half = lane >> 4;
    float bv = bf_[n];
#pragma unroll
    for (int r = 0; r < 8; ++r) {
      int row = rt * 16 + r + 8 * half;
      float resid = (row < KSL) ? qsrc_b[row * DH_ + n] : 0.f;
      o32[row * DH_ + n] = acc[r] + bv + resid;
    }
  }
  __syncthreads();

  // ---- LayerNorm per q row ----
  if (tid < 32 && tid < KSL) {
    float* orow = o32 + tid * DH_;
    float mean = 0.f;
    for (int c = 0; c < DH_; ++c) mean += orow[c];
    mean *= (1.f / (float)DH_);
    float var = 0.f;
    for (int c = 0; c < DH_; ++c) { float d = orow[c] - mean; var += d * d; }
    var *= (1.f / (float)DH_);
    float inv = rsqrtf(var + 1e-5f);
    float* dst = q_out + (size_t)b * KSL * DH_ + tid * DH_;
    for (int c = 0; c < DH_; ++c)
      dst[c] = (orow[c] - mean) * inv * ln_g[c] + ln_b[c];
  }
}

// ---------------------------------------------------------------------------
// 7) final heads: itr_mu, itr_ls, v_im = mu + exp(0.5 ls) * eps_v   (M=6400)
// ---------------------------------------------------------------------------
__global__ __launch_bounds__(256) void final_heads_kernel(
    const float* __restrict__ A, const __bf16* __restrict__ Wmu,
    const __bf16* __restrict__ Wls, const float* __restrict__ bmu,
    const float* __restrict__ bls, const float* __restrict__ eps,
    float* __restrict__ vim, float* __restrict__ muo, float* __restrict__ lso) {
  int m0 = blockIdx.x * 16;
  int wv = threadIdx.x >> 5, lane = threadIdx.x & 31;
  v16bf a[8];
#pragma unroll
  for (int kk = 0; kk < 8; ++kk) a[kk] = load_A_f32(A, DH_, m0, kk * 32, lane);
  for (int jj = 0; jj < 2; ++jj) {
    int j = wv + jj * 8;
    v8f am = zero8(), al = zero8();
#pragma unroll
    for (int kk = 0; kk < 8; ++kk) {
      am = wmma_bf(a[kk], load_B_bf(Wmu, DH_, j * 16, kk * 32, lane), am);
      al = wmma_bf(a[kk], load_B_bf(Wls, DH_, j * 16, kk * 32, lane), al);
    }
    int n = j * 16 + (lane & 15), half = lane >> 4;
    float bm = bmu[n], bl = bls[n];
#pragma unroll
    for (int r = 0; r < 8; ++r) {
      size_t row = (size_t)(m0 + r + 8 * half);
      float mu = am[r] + bm, ls = al[r] + bl;
      muo[row * DH_ + n] = mu;
      lso[row * DH_ + n] = ls;
      vim[row * DH_ + n] = mu + __expf(0.5f * ls) * eps[row * DH_ + n];
    }
  }
}

// ---------------------------------------------------------------------------
// host
// ---------------------------------------------------------------------------
extern "C" void kernel_launch(void* const* d_in, const int* in_sizes, int n_in,
                              void* d_out, int out_size, void* d_ws,
                              size_t ws_size, hipStream_t stream) {
  (void)in_sizes; (void)n_in; (void)out_size; (void)ws_size;
  const float* x_re      = (const float*)d_in[0];
  const float* v_re      = (const float*)d_in[1];
  const float* eps_x     = (const float*)d_in[2];
  const float* eps_v     = (const float*)d_in[3];
  const float* ikey_Wih  = (const float*)d_in[4];
  const float* ikey_bih  = (const float*)d_in[6];
  const float* ikey_bhh  = (const float*)d_in[7];
  const float* ikey_hl_W = (const float*)d_in[8];
  const float* ikey_hl_b = (const float*)d_in[9];
  const float* ikey_mu_W = (const float*)d_in[10];
  const float* ikey_mu_b = (const float*)d_in[11];
  const float* ikey_ls_W = (const float*)d_in[12];
  const float* ikey_ls_b = (const float*)d_in[13];
  const float* itr_Wih   = (const float*)d_in[14];
  const float* itr_bih   = (const float*)d_in[16];
  const float* itr_bhh   = (const float*)d_in[17];
  const float* aW[2][4]  = {{(const float*)d_in[18], (const float*)d_in[19],
                             (const float*)d_in[20], (const float*)d_in[21]},
                            {(const float*)d_in[25], (const float*)d_in[26],
                             (const float*)d_in[27], (const float*)d_in[28]}};
  const float* a_bf[2] = {(const float*)d_in[22], (const float*)d_in[29]};
  const float* a_g[2]  = {(const float*)d_in[23], (const float*)d_in[30]};
  const float* a_b[2]  = {(const float*)d_in[24], (const float*)d_in[31]};
  const float* itr_mu_W = (const float*)d_in[32];
  const float* itr_mu_b = (const float*)d_in[33];
  const float* itr_ls_W = (const float*)d_in[34];
  const float* itr_ls_b = (const float*)d_in[35];

  // workspace carve-out (all 256B aligned)
  char* ws = (char*)d_ws;
  size_t off = 0;
  auto carve = [&](size_t bytes) {
    char* p = ws + off;
    off += (bytes + 255) & ~(size_t)255;
    return p;
  };
  __bf16* wb_ikey_wih = (__bf16*)carve(1024 * 256 * 2);
  __bf16* wb_ikey_hl  = (__bf16*)carve(256 * 256 * 2);
  __bf16* wb_ikey_mu  = (__bf16*)carve(256 * 256 * 2);
  __bf16* wb_ikey_ls  = (__bf16*)carve(256 * 256 * 2);
  __bf16* wb_itr_wih  = (__bf16*)carve(1024 * 256 * 2);
  __bf16* wb_attn[2][4];
  for (int l = 0; l < 2; ++l) {
    wb_attn[l][0] = (__bf16*)carve(256 * 256 * 2);  // Wk
    wb_attn[l][1] = (__bf16*)carve(256 * 256 * 2);  // Wv
    wb_attn[l][2] = (__bf16*)carve(256 * 256 * 2);  // Wq
    wb_attn[l][3] = (__bf16*)carve(256 * 512 * 2);  // Wf
  }
  __bf16* wb_itr_mu = (__bf16*)carve(256 * 256 * 2);
  __bf16* wb_itr_ls = (__bf16*)carve(256 * 256 * 2);
  float* f_meanv = (float*)carve(256 * 256 * 4);
  float* f_h     = (float*)carve(256 * 256 * 4);
  float* f_hid   = (float*)carve(256 * 256 * 4);
  float* f_ikmu  = (float*)carve(256 * 256 * 4);
  float* f_iksig = (float*)carve(256 * 256 * 4);
  float* f_itrh  = (float*)carve((size_t)RSZ * 4);
  float* f_q1    = (float*)carve((size_t)RSZ * 4);
  float* f_q2    = (float*)carve((size_t)RSZ * 4);

  float* out_xim = (float*)d_out;
  float* out_vim = out_xim + RSZ;
  float* out_mu  = out_xim + 2 * (size_t)RSZ;
  float* out_ls  = out_xim + 3 * (size_t)RSZ;

  auto cvt = [&](const float* src, __bf16* dst, int rows, int src_ld, int col0,
                 int cols) {
    int total = rows * cols;
    cvt_kernel<<<(total + 255) / 256, 256, 0, stream>>>(src, dst, rows, src_ld,
                                                        col0, cols);
  };
  // weight pre-conversion (bf16), slicing out the live LSTM input columns
  cvt(ikey_Wih, wb_ikey_wih, 1024, 512, 256, 256);  // v-part only (x_prev = 0)
  cvt(ikey_hl_W, wb_ikey_hl, 256, 256, 0, 256);
  cvt(ikey_mu_W, wb_ikey_mu, 256, 256, 0, 256);
  cvt(ikey_ls_W, wb_ikey_ls, 256, 256, 0, 256);
  cvt(itr_Wih, wb_itr_wih, 1024, 512, 0, 256);      // x-part only (u_prev = 0)
  for (int l = 0; l < 2; ++l) {
    cvt(aW[l][0], wb_attn[l][0], 256, 256, 0, 256);
    cvt(aW[l][1], wb_attn[l][1], 256, 256, 0, 256);
    cvt(aW[l][2], wb_attn[l][2], 256, 256, 0, 256);
    cvt(aW[l][3], wb_attn[l][3], 256, 512, 0, 512);
  }
  cvt(itr_mu_W, wb_itr_mu, 256, 256, 0, 256);
  cvt(itr_ls_W, wb_itr_ls, 256, 256, 0, 256);

  // pipeline
  mean_v_kernel<<<NB, 256, 0, stream>>>(v_re, f_meanv);
  lstm_gates_kernel<<<256 / 16, 256, 0, stream>>>(f_meanv, wb_ikey_wih,
                                                  ikey_bih, ikey_bhh, f_h);
  gemm_relu_kernel<<<256 / 16, 256, 0, stream>>>(f_h, wb_ikey_hl, ikey_hl_b,
                                                 f_hid);
  ikey_heads_kernel<<<256 / 16, 256, 0, stream>>>(
      f_hid, wb_ikey_mu, wb_ikey_ls, ikey_mu_b, ikey_ls_b, f_ikmu, f_iksig);
  xim_kernel<<<RSZ / 256, 256, 0, stream>>>(f_ikmu, f_iksig, eps_x, out_xim);
  lstm_gates_kernel<<<(NB * KSL) / 16, 256, 0, stream>>>(
      out_xim, wb_itr_wih, itr_bih, itr_bhh, f_itrh);
  attn_kernel<<<NB, 256, 0, stream>>>(
      out_xim, x_re, v_re, out_xim, f_itrh, wb_attn[0][0], wb_attn[0][1],
      wb_attn[0][2], wb_attn[0][3], a_bf[0], a_g[0], a_b[0], f_q1);
  attn_kernel<<<NB, 256, 0, stream>>>(
      f_q1, x_re, v_re, out_xim, f_itrh, wb_attn[1][0], wb_attn[1][1],
      wb_attn[1][2], wb_attn[1][3], a_bf[1], a_g[1], a_b[1], f_q2);
  final_heads_kernel<<<(NB * KSL) / 16, 256, 0, stream>>>(
      f_q2, wb_itr_mu, wb_itr_ls, itr_mu_b, itr_ls_b, eps_v, out_vim, out_mu,
      out_ls);
}
